// MyModel_57088705299065
// MI455X (gfx1250) — matrix operand order repaired and checked by reading, not compiled, not run on hardware
//
#include <hip/hip_runtime.h>
#include <hip/hip_bf16.h>
#include <math.h>
#include <stdint.h>

#define VOCABN   50000
#define EMBN     300
#define EMB_PAD  320
#define SEQN     256
#define HIDN     256
#define TAGSN    50
#define TAGS_PAD 64
#define BATCHN   128
#define G4H      1024            // 4*HID
#define M_ALL    (SEQN * BATCHN) // 32768 rows (index = s*BATCH + b)

typedef __bf16 bf16;
typedef __attribute__((ext_vector_type(16))) __bf16 v16bf;
typedef __attribute__((ext_vector_type(8)))  __bf16 v8bf;
typedef __attribute__((ext_vector_type(8)))  float  v8f;

// ---------------------------------------------------------------------------
// CDNA5 async copy: global -> LDS, 16 bytes per lane, tracked by ASYNCcnt.
// (CDNA5 ISA; inline asm per bridge-doc guidance)
// ---------------------------------------------------------------------------
__device__ __forceinline__ void async_ld_lds16(uint32_t lds_off, uint64_t gaddr) {
    asm volatile("global_load_async_to_lds_b128 %0, %1, off"
                 :: "v"(lds_off), "v"(gaddr) : "memory");
}
__device__ __forceinline__ void wait_async_zero() {
    asm volatile("s_wait_asynccnt 0x0" ::: "memory");
}

// Stage a contiguous [64 x ldb] bf16 strip of BT into LDS (row-major, same layout).
// bytes = 64*ldb*2, always a multiple of blockDim.x*16 for our shapes.
__device__ __forceinline__ void stage_B_strip(char* smem, const bf16* __restrict__ BT,
                                              int ldb, int n0) {
    const uint64_t src = (uint64_t)(const char*)(BT + (size_t)n0 * ldb);
    const uint32_t dst = (uint32_t)(uintptr_t)smem;  // LDS addr = generic[31:0]
    const int nchunks = (64 * ldb * 2) / 16;         // 16B chunks
    for (int i = threadIdx.x; i < nchunks; i += blockDim.x) {
        async_ld_lds16(dst + (uint32_t)i * 16u, src + (uint64_t)i * 16u);
    }
    wait_async_zero();
    __syncthreads();
}

// ---------------------------------------------------------------------------
// WMMA operand loaders (contiguous k order, same for A and B; not the ISA table order)
// A: 16x32 (MxK). lane l: g=l>>4, m=l&15;
//    halves 0..15 -> K=k0+16g+i
// ---------------------------------------------------------------------------
__device__ __forceinline__ v16bf load_A_tile(const bf16* __restrict__ A, int lda,
                                             int m0, int k0) {
    int lane = threadIdx.x & 31;
    int g = lane >> 4, m = lane & 15;
    const bf16* p = A + (size_t)(m0 + m) * lda + k0 + 16 * g;
    v8bf lo = *(const v8bf*)(p);
    v8bf hi = *(const v8bf*)(p + 8);
    return __builtin_shufflevector(lo, hi, 0,1,2,3,4,5,6,7,8,9,10,11,12,13,14,15);
}

// B tile from the LDS-resident strip [64][ldb]: lane l: n=l&15, K=k0+16*(l>>4)+i
__device__ __forceinline__ v16bf load_B_lds(const bf16* sm, int ldb, int nloc, int k0) {
    int lane = threadIdx.x & 31;
    int g = lane >> 4, n = lane & 15;
    const bf16* p = sm + (size_t)(nloc + n) * ldb + k0 + 16 * g;
    return *(const v16bf*)p;
}

// C/D f32 16x16: lane l, vgpr r: m = r + 8*(l>>4), n = l&15
__device__ __forceinline__ void store_C_tile(float* __restrict__ C, int ldc,
                                             int m0, int n0, v8f acc,
                                             const float* __restrict__ bias, int biasN,
                                             const float* __restrict__ Cadd) {
    int lane = threadIdx.x & 31;
    int n = lane & 15, half = lane >> 4;
    int col = n0 + n;
    float bv = bias ? bias[col < biasN ? col : (biasN - 1)] : 0.f;
#pragma unroll
    for (int r = 0; r < 8; ++r) {
        int row = m0 + r + 8 * half;
        float v = acc[r] + bv;
        if (Cadd) v += Cadd[(size_t)row * ldc + col];
        C[(size_t)row * ldc + col] = v;
    }
}

#define WMMA_BF16(a, b, c) \
    __builtin_amdgcn_wmma_f32_16x16x32_bf16(false, (a), false, (b), (short)0, (c), false, false)

// ---------------------------------------------------------------------------
// LDS-staged bf16 WMMA GEMM. Block = 8 waves = 128(M) x 64(N) macro-tile.
// B strip (64 x K) async-copied to LDS once per block; each wave then runs
// 4 WMMAs per K-step against its own A tile. M multiple of 128, N of 64.
// All tile choices are wave-uniform (EXEC all-1s for WMMA).
// ---------------------------------------------------------------------------
__global__ void wmma_gemm_lds_kernel(const bf16* __restrict__ A, int lda,
                                     const bf16* __restrict__ BT, int ldb,
                                     const float* __restrict__ bias, int biasN,
                                     const float* __restrict__ Cadd,
                                     float* __restrict__ C, int ldc,
                                     int Ngroups, int K) {
    extern __shared__ char smem[];
    int mb = blockIdx.x / Ngroups;
    int ng = blockIdx.x % Ngroups;
    int n0 = ng * 64;

    stage_B_strip(smem, BT, ldb, n0);
    const bf16* sB = (const bf16*)smem;

    int wave = threadIdx.x >> 5;
    int m0 = mb * 128 + wave * 16;

    v8f acc0 = {}, acc1 = {}, acc2 = {}, acc3 = {};
    for (int k0 = 0; k0 < K; k0 += 32) {
        if (k0 + 32 < K) {
            int lane = threadIdx.x & 31;
            __builtin_prefetch(A + (size_t)(m0 + (lane & 15)) * lda + k0 + 32, 0, 1);
        }
        v16bf a  = load_A_tile(A, lda, m0, k0);
        v16bf b0 = load_B_lds(sB, ldb,  0, k0);
        v16bf b1 = load_B_lds(sB, ldb, 16, k0);
        v16bf b2 = load_B_lds(sB, ldb, 32, k0);
        v16bf b3 = load_B_lds(sB, ldb, 48, k0);
        acc0 = WMMA_BF16(a, b0, acc0);
        acc1 = WMMA_BF16(a, b1, acc1);
        acc2 = WMMA_BF16(a, b2, acc2);
        acc3 = WMMA_BF16(a, b3, acc3);
    }
    store_C_tile(C, ldc, m0, n0 +  0, acc0, bias, biasN, Cadd);
    store_C_tile(C, ldc, m0, n0 + 16, acc1, bias, biasN, Cadd);
    store_C_tile(C, ldc, m0, n0 + 32, acc2, bias, biasN, Cadd);
    store_C_tile(C, ldc, m0, n0 + 48, acc3, bias, biasN, Cadd);
}

// ---------------------------------------------------------------------------
// Recurrent step GEMM: both directions, z = xz[dir][t] + h[dir] @ Wh[dir]
// M=128, N=1024, K=256 fully unrolled. Grid: 2 dirs x 16 Ngroups = 32 blocks.
// Wh strip (64x256 bf16 = 32KB) async-staged into LDS per block.
// ---------------------------------------------------------------------------
__global__ void step_gemm_kernel(const bf16* __restrict__ hbf,    // [2][128][256]
                                 const bf16* __restrict__ WhT_f,  // [1024][256]
                                 const bf16* __restrict__ WhT_b,
                                 const float* __restrict__ xz_f,  // [S][128][1024]
                                 const float* __restrict__ xz_b,
                                 float* __restrict__ zbuf,        // [2][128][1024]
                                 int it) {
    extern __shared__ char smem[];
    int dir = blockIdx.x >> 4;       // /16
    int ng  = blockIdx.x & 15;
    int n0 = ng * 64;
    int t = dir ? (SEQN - 1 - it) : it;

    const bf16* A  = hbf + (size_t)dir * BATCHN * HIDN;
    const bf16* BT = dir ? WhT_b : WhT_f;
    const float* Cadd = (dir ? xz_b : xz_f) + (size_t)t * BATCHN * G4H;
    float* C = zbuf + (size_t)dir * BATCHN * G4H;

    stage_B_strip(smem, BT, HIDN, n0);
    const bf16* sB = (const bf16*)smem;

    int wave = threadIdx.x >> 5;
    int m0 = wave * 16;

    v8f acc0 = {}, acc1 = {}, acc2 = {}, acc3 = {};
#pragma unroll
    for (int k0 = 0; k0 < HIDN; k0 += 32) {
        v16bf a  = load_A_tile(A, HIDN, m0, k0);
        v16bf b0 = load_B_lds(sB, HIDN,  0, k0);
        v16bf b1 = load_B_lds(sB, HIDN, 16, k0);
        v16bf b2 = load_B_lds(sB, HIDN, 32, k0);
        v16bf b3 = load_B_lds(sB, HIDN, 48, k0);
        acc0 = WMMA_BF16(a, b0, acc0);
        acc1 = WMMA_BF16(a, b1, acc1);
        acc2 = WMMA_BF16(a, b2, acc2);
        acc3 = WMMA_BF16(a, b3, acc3);
    }
    store_C_tile(C, G4H, m0, n0 +  0, acc0, nullptr, 0, Cadd);
    store_C_tile(C, G4H, m0, n0 + 16, acc1, nullptr, 0, Cadd);
    store_C_tile(C, G4H, m0, n0 + 32, acc2, nullptr, 0, Cadd);
    store_C_tile(C, G4H, m0, n0 + 48, acc3, nullptr, 0, Cadd);
}

// ---------------------------------------------------------------------------
// Elementwise LSTM gates (Keras order i,f,g,o; activation=relu):
// c = sig(f)*c + sig(i)*relu(g);  h = sig(o)*relu(c)
// ---------------------------------------------------------------------------
__device__ __forceinline__ float sigf(float x) { return 1.f / (1.f + __expf(-x)); }

__global__ void gate_update_kernel(const float* __restrict__ zbuf, // [2][128][1024]
                                   float* __restrict__ c_state,    // [2][128][256]
                                   bf16* __restrict__ hbf,         // [2][128][256]
                                   bf16* __restrict__ hs,          // [S][128][512]
                                   int it) {
    int idx = blockIdx.x * blockDim.x + threadIdx.x;
    if (idx >= 2 * BATCHN * HIDN) return;
    int dir = idx / (BATCHN * HIDN);
    int rem = idx % (BATCHN * HIDN);
    int b = rem / HIDN, j = rem % HIDN;

    const float* z = zbuf + (size_t)dir * BATCHN * G4H + (size_t)b * G4H;
    float zi = z[j], zf = z[HIDN + j], zg = z[2 * HIDN + j], zo = z[3 * HIDN + j];
    float c = c_state[idx];
    c = sigf(zf) * c + sigf(zi) * fmaxf(zg, 0.f);
    c_state[idx] = c;
    float h = sigf(zo) * fmaxf(c, 0.f);
    hbf[idx] = (bf16)h;
    int t = dir ? (SEQN - 1 - it) : it;
    hs[((size_t)t * BATCHN + b) * (2 * HIDN) + dir * HIDN + j] = (bf16)h;
}

// ---------------------------------------------------------------------------
// Prep kernels
// ---------------------------------------------------------------------------
// W[K][N] f32 -> WT[Npad][Kpad] bf16 (zero-padded)
__global__ void transpose_pad_kernel(const float* __restrict__ W, bf16* __restrict__ WT,
                                     int K, int N, int Kpad, int Npad) {
    int id = blockIdx.x * blockDim.x + threadIdx.x;
    if (id >= Npad * Kpad) return;
    int n = id / Kpad, k = id % Kpad;
    float v = (k < K && n < N) ? W[(size_t)k * N + n] : 0.f;
    WT[id] = (bf16)v;
}

// embedding gather -> x_bf16 [S*B][EMB_PAD], row = s*B + b
__global__ void embed_kernel(const int* __restrict__ tokens,
                             const float* __restrict__ emb,
                             bf16* __restrict__ x) {
    long id = (long)blockIdx.x * blockDim.x + threadIdx.x;
    if (id >= (long)M_ALL * EMB_PAD) return;
    int row = (int)(id / EMB_PAD);
    int k   = (int)(id % EMB_PAD);
    int s = row / BATCHN, b = row % BATCHN;
    float v = 0.f;
    if (k < EMBN) {
        int tok = tokens[(size_t)b * SEQN + s];
        v = emb[(size_t)tok * EMBN + k];
    }
    x[id] = (bf16)v;
}

__global__ void zero_state_kernel(float* __restrict__ c_state, bf16* __restrict__ hbf) {
    int idx = blockIdx.x * blockDim.x + threadIdx.x;
    if (idx >= 2 * BATCHN * HIDN) return;
    c_state[idx] = 0.f;
    hbf[idx] = (bf16)0.f;
}

// ---------------------------------------------------------------------------
// Softmax over TAGS per row; one wave per row; output reordered to [B][S][TAGS]
// ---------------------------------------------------------------------------
__global__ void softmax_kernel(const float* __restrict__ logits, // [S*B][TAGS_PAD]
                               float* __restrict__ out) {        // [B][S][TAGS]
    int wave = threadIdx.x >> 5;
    int lane = threadIdx.x & 31;
    int row = blockIdx.x * (blockDim.x >> 5) + wave;
    if (row >= M_ALL) return;
    const float* lr = logits + (size_t)row * TAGS_PAD;
    float v0 = (lane < TAGSN) ? lr[lane] : -1e30f;
    float v1 = (lane + 32 < TAGSN) ? lr[lane + 32] : -1e30f;
    float m = fmaxf(v0, v1);
#pragma unroll
    for (int off = 16; off; off >>= 1) m = fmaxf(m, __shfl_xor(m, off, 32));
    float e0 = (lane < TAGSN) ? __expf(v0 - m) : 0.f;
    float e1 = (lane + 32 < TAGSN) ? __expf(v1 - m) : 0.f;
    float s = e0 + e1;
#pragma unroll
    for (int off = 16; off; off >>= 1) s += __shfl_xor(s, off, 32);
    float inv = 1.f / s;
    int b = row % BATCHN, t = row / BATCHN;
    float* o = out + ((size_t)b * SEQN + t) * TAGSN;
    if (lane < TAGSN) o[lane] = e0 * inv;
    if (lane + 32 < TAGSN) o[lane + 32] = e1 * inv;
}

// ---------------------------------------------------------------------------
// host side
// ---------------------------------------------------------------------------
extern "C" void kernel_launch(void* const* d_in, const int* in_sizes, int n_in,
                              void* d_out, int out_size, void* d_ws, size_t ws_size,
                              hipStream_t stream) {
    (void)in_sizes; (void)n_in; (void)out_size; (void)ws_size;
    const int*   tokens = (const int*)  d_in[0];
    const float* emb    = (const float*)d_in[1];
    const float* Wx_f   = (const float*)d_in[2];
    const float* Wh_f   = (const float*)d_in[3];
    const float* b_f    = (const float*)d_in[4];
    const float* Wx_b   = (const float*)d_in[5];
    const float* Wh_b   = (const float*)d_in[6];
    const float* b_b    = (const float*)d_in[7];
    const float* Wd     = (const float*)d_in[8];
    const float* bd     = (const float*)d_in[9];
    float* out = (float*)d_out;

    char* ws = (char*)d_ws;
    size_t off = 0;
    auto carve = [&](size_t bytes) -> void* {
        void* p = ws + off;
        off = (off + bytes + 255) & ~(size_t)255;
        return p;
    };
    bf16*  xbf    = (bf16*) carve((size_t)M_ALL * EMB_PAD * sizeof(bf16));
    bf16*  WxTf   = (bf16*) carve((size_t)G4H * EMB_PAD * sizeof(bf16));
    bf16*  WxTb   = (bf16*) carve((size_t)G4H * EMB_PAD * sizeof(bf16));
    bf16*  WhTf   = (bf16*) carve((size_t)G4H * HIDN * sizeof(bf16));
    bf16*  WhTb   = (bf16*) carve((size_t)G4H * HIDN * sizeof(bf16));
    bf16*  WdT    = (bf16*) carve((size_t)TAGS_PAD * 2 * HIDN * sizeof(bf16));
    float* xz_f   = (float*)carve((size_t)M_ALL * G4H * sizeof(float));
    float* xz_b   = (float*)carve((size_t)M_ALL * G4H * sizeof(float));
    float* zbuf   = (float*)carve((size_t)2 * BATCHN * G4H * sizeof(float));
    float* cstate = (float*)carve((size_t)2 * BATCHN * HIDN * sizeof(float));
    bf16*  hbf    = (bf16*) carve((size_t)2 * BATCHN * HIDN * sizeof(bf16));
    bf16*  hs     = (bf16*) carve((size_t)M_ALL * 2 * HIDN * sizeof(bf16));
    float* logits = (float*)carve((size_t)M_ALL * TAGS_PAD * sizeof(float));

    const int TPB = 256;

    // --- weight prep (transpose + bf16 cast + pad) ---
    transpose_pad_kernel<<<(G4H * EMB_PAD + TPB - 1) / TPB, TPB, 0, stream>>>(
        Wx_f, WxTf, EMBN, G4H, EMB_PAD, G4H);
    transpose_pad_kernel<<<(G4H * EMB_PAD + TPB - 1) / TPB, TPB, 0, stream>>>(
        Wx_b, WxTb, EMBN, G4H, EMB_PAD, G4H);
    transpose_pad_kernel<<<(G4H * HIDN + TPB - 1) / TPB, TPB, 0, stream>>>(
        Wh_f, WhTf, HIDN, G4H, HIDN, G4H);
    transpose_pad_kernel<<<(G4H * HIDN + TPB - 1) / TPB, TPB, 0, stream>>>(
        Wh_b, WhTb, HIDN, G4H, HIDN, G4H);
    transpose_pad_kernel<<<(TAGS_PAD * 2 * HIDN + TPB - 1) / TPB, TPB, 0, stream>>>(
        Wd, WdT, 2 * HIDN, TAGSN, 2 * HIDN, TAGS_PAD);

    // --- embedding gather -> bf16, padded ---
    {
        long total = (long)M_ALL * EMB_PAD;
        embed_kernel<<<(int)((total + TPB - 1) / TPB), TPB, 0, stream>>>(tokens, emb, xbf);
    }

    // --- zero recurrent state ---
    zero_state_kernel<<<(2 * BATCHN * HIDN + TPB - 1) / TPB, TPB, 0, stream>>>(cstate, hbf);

    // --- input projections: xz = x @ Wx + b  (both directions) ---
    {
        int Mblocks = M_ALL / 128, Ngroups = G4H / 64;        // 256 x 16
        int blocks = Mblocks * Ngroups;                       // 4096
        size_t shmem = (size_t)64 * EMB_PAD * sizeof(bf16);   // 40 KB
        wmma_gemm_lds_kernel<<<blocks, TPB, shmem, stream>>>(
            xbf, EMB_PAD, WxTf, EMB_PAD, b_f, G4H, nullptr, xz_f, G4H, Ngroups, EMB_PAD);
        wmma_gemm_lds_kernel<<<blocks, TPB, shmem, stream>>>(
            xbf, EMB_PAD, WxTb, EMB_PAD, b_b, G4H, nullptr, xz_b, G4H, Ngroups, EMB_PAD);
    }

    // --- sequential recurrence, both directions fused per step ---
    {
        size_t shmem = (size_t)64 * HIDN * sizeof(bf16);      // 32 KB
        for (int it = 0; it < SEQN; ++it) {
            step_gemm_kernel<<<32, TPB, shmem, stream>>>(hbf, WhTf, WhTb, xz_f, xz_b, zbuf, it);
            gate_update_kernel<<<(2 * BATCHN * HIDN + TPB - 1) / TPB, TPB, 0, stream>>>(
                zbuf, cstate, hbf, hs, it);
        }
    }

    // --- dense head: logits = hs @ Wd + bd ---
    {
        int Mblocks = M_ALL / 128, Ngroups = TAGS_PAD / 64;   // 256 x 1
        size_t shmem = (size_t)64 * 2 * HIDN * sizeof(bf16);  // 64 KB
        wmma_gemm_lds_kernel<<<Mblocks * Ngroups, TPB, shmem, stream>>>(
            hs, 2 * HIDN, WdT, 2 * HIDN, bd, TAGSN, nullptr, logits, TAGS_PAD,
            Ngroups, 2 * HIDN);
    }

    // --- softmax + reorder to [B][S][TAGS] ---
    softmax_kernel<<<M_ALL / (TPB / 32), TPB, 0, stream>>>(logits, out);
}
